// RRPlus_M5_48515950576136
// MI455X (gfx1250) — matrix operand order repaired and checked by reading, not compiled, not run on hardware
//
#include <hip/hip_runtime.h>
#include <hip/hip_bf16.h>
#include <math.h>

// ---------------------------------------------------------------------------
// Types for WMMA fragments / TDM descriptors (gfx1250, wave32)
// ---------------------------------------------------------------------------
typedef __attribute__((ext_vector_type(16))) __bf16 v16bf;
typedef __attribute__((ext_vector_type(8)))  float  v8f;
typedef __attribute__((ext_vector_type(4)))  unsigned int u32x4;
typedef __attribute__((ext_vector_type(4)))  unsigned int v4u;
typedef __attribute__((ext_vector_type(4)))  int v4i;
typedef __attribute__((ext_vector_type(8)))  int v8i;

union FragU {
    u32x4 u[2];
    v16bf v;
};

#define AS3 __attribute__((address_space(3)))

// LDS row stride (in bf16 elements) for A/B slabs: 40*2B = 80B, multiple of 16B.
#define AS 40

// Raw LDS byte offset of a __shared__ object (generic -> local ptrtoint).
__device__ __forceinline__ unsigned lds_addr_of(const void* p) {
    return (unsigned)(unsigned long long)(const AS3 void*)p;
}

// ---------------------------------------------------------------------------
// TDM: async 2D tile load global->LDS via Tensor Data Mover.
//   Loads `rows` rows of 32 bf16 elements, source row stride `kpad` elements,
//   into LDS at `ldsoff` with 80B destination row stride (pad 16B per 64B row).
// Tracked by TENSORcnt.
// ---------------------------------------------------------------------------
__device__ __forceinline__ void tdm_load_A(const __bf16* gsrc, unsigned ldsoff,
                                           int rows, int kpad) {
    unsigned long long ga = (unsigned long long)(const void*)gsrc;
    v4u g0;
    g0[0] = 1u;                                        // count=1, user mode
    g0[1] = ldsoff;                                    // lds_addr (bytes)
    g0[2] = (unsigned)(ga & 0xffffffffull);            // global_addr[31:0]
    g0[3] = (unsigned)((ga >> 32) & 0x1ffffffull)      // global_addr[56:32]
          | 0x80000000u;                               // type = 2 ("image")
    const unsigned dim0 = 32u;                         // tensor/tile dim0 (elems)
    const unsigned dim1 = (unsigned)rows;
    const unsigned long long d0s = (unsigned long long)kpad;  // row stride
    v8i g1;
    g1[0] = (int)((1u << 16)       // data_size = 2 bytes
                | (1u << 20)       // pad_enable
                | (3u << 22)       // pad_interval: every 64B
                | (3u << 25));     // pad_amount: 4 DWORDs (16B) -> 80B stride
    g1[1] = (int)((dim0 & 0xffffu) << 16);             // tensor_dim0[15:0]
    g1[2] = (int)((dim0 >> 16) | ((dim1 & 0xffffu) << 16));
    g1[3] = (int)((dim1 >> 16) | (dim0 << 16));        // tile_dim0 = 32
    g1[4] = (int)(dim1 & 0xffffu);                     // tile_dim1 = rows, tile_dim2=0
    g1[5] = (int)(d0s & 0xffffffffull);                // tensor_dim0_stride lo
    g1[6] = (int)((d0s >> 32) & 0xffffull);            // stride hi | dim1_stride lo(0)
    g1[7] = 0;
    v4i z4 = {0, 0, 0, 0};
#if __clang_major__ >= 23
    v8i z8 = {0, 0, 0, 0, 0, 0, 0, 0};
    __builtin_amdgcn_tensor_load_to_lds(g0, g1, z4, z4, z8, 0);
#else
    __builtin_amdgcn_tensor_load_to_lds(g0, g1, z4, z4, 0);
#endif
}

// ---------------------------------------------------------------------------
// Weight prep: fp32 [Cout][Kin] -> padded bf16 [Mpad][Kpad]
// ---------------------------------------------------------------------------
__global__ __launch_bounds__(256) void prep_w(const float* __restrict__ w,
                                              __bf16* __restrict__ Wa,
                                              int Cout, int Kin, int Mpad, int Kpad) {
    int idx = blockIdx.x * 256 + threadIdx.x;
    if (idx >= Mpad * Kpad) return;
    int m = idx / Kpad, k = idx - m * Kpad;
    float v = (m < Cout && k < Kin) ? w[m * Kin + k] : 0.0f;
    Wa[idx] = (__bf16)v;
}

// ---------------------------------------------------------------------------
// Lift layer (Cin=1, K=79 dilated conv) + fused maxpool4, via WMMA.
//   x  : [8, 16000] fp32
//   Wa : [80][96] bf16 (padded w1)
//   y1 : [8, 74, 9, 4000] fp32  (pooled, scaled by 1/2^j)
// grid: (125 nblocks, 9 scales, 8 batch), block 256 (8 waves)
// A slab double-buffered via TDM; each wave owns one 16-wide n-tile.
// ---------------------------------------------------------------------------
__global__ __launch_bounds__(256) void lift_wmma(const float* __restrict__ x,
                                                 const __bf16* __restrict__ Wa,
                                                 float* __restrict__ y1) {
    constexpr int MT = 5, KT = 3, KPAD = 96, L = 16000, COUT = 74;
    __shared__ __align__(16) __bf16 Alds[2][MT * 16 * AS];   // 2 x 80 rows
    __shared__ __align__(16) __bf16 Blds[128 * AS];
    __shared__ __align__(16) float  Cbuf[80 * 132];

    const int nb  = blockIdx.x;
    const int j   = blockIdx.y;
    const int b   = blockIdx.z;
    const int t0  = nb * 128;
    const int dil = 1 << j;
    const int tid  = threadIdx.x;
    const int lane = tid & 31;
    const int wave = tid >> 5;          // 0..7 : n-tile
    const int l16  = lane & 15;
    const int hi   = lane >> 4;
    const float* xb = x + (size_t)b * L;
    const unsigned a_base  = lds_addr_of(&Alds[0][0]);
    const unsigned a_bytes = (unsigned)(MT * 16 * AS * 2);

    v8f acc[MT];
#pragma unroll
    for (int q = 0; q < MT; ++q)
#pragma unroll
        for (int e = 0; e < 8; ++e) acc[q][e] = 0.0f;

    if (wave == 0) tdm_load_A(Wa, a_base, MT * 16, KPAD);   // A chunk kt=0

    for (int kt = 0; kt < KT; ++kt) {
        const int k0 = kt * 32;
        // ---- fill B slab (im2col): Blds[n][kk] = x[t0+n + (kg-39)*dil] ----
        for (int idx = tid; idx < 32 * 128; idx += 256) {
            int kk = idx >> 7, n = idx & 127;
            int kg = k0 + kk;
            __bf16 val = (__bf16)0.0f;
            if (kg < 79) {
                int t = t0 + n + (kg - 39) * dil;
                if (t >= 0 && t < L) val = (__bf16)xb[t];
            }
            Blds[n * AS + kk] = val;
        }
        if (wave == 0) __builtin_amdgcn_s_wait_tensorcnt(0);  // A(kt) landed
        __syncthreads();
        // prefetch next A chunk into the alternate buffer, overlaps compute
        if (wave == 0 && kt + 1 < KT)
            tdm_load_A(Wa + (kt + 1) * 32, a_base + ((kt + 1) & 1) * a_bytes,
                       MT * 16, KPAD);
        const __bf16* Acur = &Alds[kt & 1][0];
        // ---- B fragment for this wave's n-tile ----
        FragU bf;
        const __bf16* brow = Blds + (wave * 16 + l16) * AS + hi * 8;
        bf.u[0] = *(const u32x4*)(brow);
        bf.u[1] = *(const u32x4*)(brow + 16);
        // ---- WMMA over all m-tiles ----
#pragma unroll
        for (int mt = 0; mt < MT; ++mt) {
            FragU af;
            const __bf16* arow = Acur + (mt * 16 + l16) * AS;
            af.u[0] = *(const u32x4*)(arow + 8 * hi);
            af.u[1] = *(const u32x4*)(arow + 16 + 8 * hi);
            acc[mt] = __builtin_amdgcn_wmma_f32_16x16x32_bf16(
                false, af.v, false, bf.v, (short)0, acc[mt], false, false);
        }
        __syncthreads();
    }
    // ---- stage conv tile to LDS, then pool4 + scale + store ----
#pragma unroll
    for (int mt = 0; mt < MT; ++mt)
#pragma unroll
        for (int v = 0; v < 8; ++v)
            Cbuf[(mt * 16 + v + 8 * hi) * 132 + wave * 16 + l16] = acc[mt][v];
    __syncthreads();
    const float sc = 1.0f / (float)dil;
    const int tp0 = nb * 32;
    for (int idx = tid; idx < COUT * 32; idx += 256) {
        int co = idx >> 5, tp = idx & 31;
        const float* r = Cbuf + co * 132 + tp * 4;
        float m4 = fmaxf(fmaxf(r[0], r[1]), fmaxf(r[2], r[3]));
        y1[(((size_t)b * COUT + co) * 9 + j) * 4000 + tp0 + tp] = m4 * sc;
    }
}

// ---------------------------------------------------------------------------
// Group conv (valid over scale axis, dilated over time) via WMMA + TDM.
//   yin : [8, CIN, HIN, L] bf16       Wa : [MTpad*16][KT*32] bf16
//   yout: [8, COUT, HIN-2, L] fp32 (scaled 1/2^i)
// grid: (ceil(L/64), HIN-2, 8), block 256.
// waves: ntile = wave&3 (4 x 16 N), mgrp = wave>>2 (m-tiles strided by 2).
// ---------------------------------------------------------------------------
template <int COUT, int CIN, int HIN, int L>
__global__ __launch_bounds__(256) void gconv_wmma(const __bf16* __restrict__ yin,
                                                  const __bf16* __restrict__ Wa,
                                                  float* __restrict__ yout) {
    constexpr int MT   = (COUT + 15) / 16;
    constexpr int K    = CIN * 9;
    constexpr int KT   = (K + 31) / 32;
    constexpr int KPAD = KT * 32;
    constexpr int HOUT = HIN - 2;
    constexpr int MAXQ = (MT + 1) / 2;
    __shared__ __align__(16) __bf16 Alds[2][MT * 16 * AS];
    __shared__ __align__(16) __bf16 Blds[64 * AS];

    const int nb  = blockIdx.x;
    const int i   = blockIdx.y;          // output scale
    const int b   = blockIdx.z;
    const int t0  = nb * 64;
    const int dil = 1 << i;
    const int tid   = threadIdx.x;
    const int lane  = tid & 31;
    const int wave  = tid >> 5;
    const int l16   = lane & 15;
    const int hi    = lane >> 4;
    const int ntile = wave & 3;
    const int mgrp  = wave >> 2;
    const __bf16* yb = yin + (size_t)b * CIN * HIN * L;
    const unsigned a_base  = lds_addr_of(&Alds[0][0]);
    const unsigned a_bytes = (unsigned)(MT * 16 * AS * 2);

    v8f acc[MAXQ];
#pragma unroll
    for (int q = 0; q < MAXQ; ++q)
#pragma unroll
        for (int e = 0; e < 8; ++e) acc[q][e] = 0.0f;

    if (wave == 0) tdm_load_A(Wa, a_base, MT * 16, KPAD);    // A chunk kt=0

    for (int kt = 0; kt < KT; ++kt) {
        const int k0 = kt * 32;
        if (kt + 1 < KT) __builtin_prefetch((const void*)(Wa + (kt + 1) * 32), 0, 0);
        // ---- fill B slab (im2col): K = (ci, dh, dk) ----
        for (int idx = tid; idx < 32 * 64; idx += 256) {
            int kk = idx >> 6, n = idx & 63;
            int kg = k0 + kk;
            __bf16 val = (__bf16)0.0f;
            if (kg < K) {
                int ci = kg / 9, r = kg - ci * 9;
                int dh = r / 3, dk = r - dh * 3;
                int t = t0 + n + (dk - 1) * dil;
                if (t >= 0 && t < L)
                    val = yb[((size_t)ci * HIN + (i + dh)) * L + t];
            }
            Blds[n * AS + kk] = val;
        }
        if (wave == 0) __builtin_amdgcn_s_wait_tensorcnt(0);  // A(kt) landed
        __syncthreads();
        // prefetch next A chunk into the alternate buffer, overlaps compute
        if (wave == 0 && kt + 1 < KT)
            tdm_load_A(Wa + (size_t)(kt + 1) * 32, a_base + ((kt + 1) & 1) * a_bytes,
                       MT * 16, KPAD);
        const __bf16* Acur = &Alds[kt & 1][0];
        FragU bf;
        const __bf16* brow = Blds + (ntile * 16 + l16) * AS + hi * 8;
        bf.u[0] = *(const u32x4*)(brow);
        bf.u[1] = *(const u32x4*)(brow + 16);
        int q = 0;
        for (int mt = mgrp; mt < MT; mt += 2, ++q) {
            FragU af;
            const __bf16* arow = Acur + (mt * 16 + l16) * AS;
            af.u[0] = *(const u32x4*)(arow + 8 * hi);
            af.u[1] = *(const u32x4*)(arow + 16 + 8 * hi);
            acc[q] = __builtin_amdgcn_wmma_f32_16x16x32_bf16(
                false, af.v, false, bf.v, (short)0, acc[q], false, false);
        }
        __syncthreads();
    }
    // ---- epilogue: scale by 1/2^i, bounds-checked store ----
    const float sc = 1.0f / (float)dil;
    const int t = t0 + ntile * 16 + l16;
    if (t < L) {
        int q = 0;
        for (int mt = mgrp; mt < MT; mt += 2, ++q) {
#pragma unroll
            for (int v = 0; v < 8; ++v) {
                int co = mt * 16 + v + 8 * hi;
                if (co < COUT)
                    yout[(((size_t)b * COUT + co) * HOUT + i) * L + t] = acc[q][v] * sc;
            }
        }
    }
}

// ---------------------------------------------------------------------------
// BatchNorm batch statistics: one block per channel, deterministic tree reduce.
//   y: [8, C, HL] fp32 -> mean[c], rstd[c]
// ---------------------------------------------------------------------------
__global__ __launch_bounds__(256) void bn_stats(const float* __restrict__ y,
                                                float* __restrict__ mean,
                                                float* __restrict__ rstd,
                                                int C, int HL) {
    int c = blockIdx.x;
    double s = 0.0, q = 0.0;
    for (int idx = threadIdx.x; idx < 8 * HL; idx += 256) {
        int b = idx / HL, r = idx - b * HL;
        float v = y[((size_t)b * C + c) * HL + r];
        s += (double)v;
        q += (double)v * (double)v;
    }
    __shared__ double rs[256], rq[256];
    rs[threadIdx.x] = s;
    rq[threadIdx.x] = q;
    __syncthreads();
    for (int off = 128; off > 0; off >>= 1) {
        if (threadIdx.x < (unsigned)off) {
            rs[threadIdx.x] += rs[threadIdx.x + off];
            rq[threadIdx.x] += rq[threadIdx.x + off];
        }
        __syncthreads();
    }
    if (threadIdx.x == 0) {
        double n  = (double)(8 * HL);
        double mu = rs[0] / n;
        double var = rq[0] / n - mu * mu;
        mean[c] = (float)mu;
        rstd[c] = (float)(1.0 / sqrt(var + 2e-5));
    }
}

// ---------------------------------------------------------------------------
// BN (batch stats) + ReLU + maxpool4, elementwise.
//   yin: [8, C, H, L] fp32 -> yout: [8, C, H, L/4] (bf16 or fp32)
// ---------------------------------------------------------------------------
template <bool BF16OUT>
__global__ __launch_bounds__(256) void bn_relu_pool(const float* __restrict__ yin,
                                                    const float* __restrict__ mean,
                                                    const float* __restrict__ rstd,
                                                    const float* __restrict__ g,
                                                    const float* __restrict__ bb,
                                                    void* __restrict__ yout,
                                                    int C, int H, int L, int Lo) {
    int idx = blockIdx.x * 256 + threadIdx.x;
    int total = 8 * C * H * Lo;
    if (idx >= total) return;
    int tp = idx % Lo;
    int r = idx / Lo;
    int h = r % H; r /= H;
    int c = r % C;
    int b = r / C;
    const float* p = yin + (((size_t)b * C + c) * H + h) * L + tp * 4;
    float sc  = g[c] * rstd[c];
    float off = bb[c] - mean[c] * sc;
    float v = fmaxf(fmaxf(p[0] * sc + off, p[1] * sc + off),
                    fmaxf(p[2] * sc + off, p[3] * sc + off));
    v = fmaxf(v, 0.0f);
    if (BF16OUT) ((__bf16*)yout)[idx] = (__bf16)v;
    else         ((float*)yout)[idx] = v;
}

// ---------------------------------------------------------------------------
// Global average over time: yin [n][T] -> yout [n]
// ---------------------------------------------------------------------------
__global__ __launch_bounds__(256) void mean_time(const float* __restrict__ yin,
                                                 float* __restrict__ yout,
                                                 int n, int T) {
    int idx = blockIdx.x * 256 + threadIdx.x;
    if (idx >= n) return;
    float s = 0.0f;
    for (int t = 0; t < T; ++t) s += yin[(size_t)idx * T + t];
    yout[idx] = s / (float)T;
}

// ---------------------------------------------------------------------------
// Final head: gconv with w5 [10,296,3,1] over y8 [8,296,3]; H collapses to 1,
// max over scale axis of a single scale == identity. out: [8,10] fp32.
// ---------------------------------------------------------------------------
__global__ __launch_bounds__(128) void final_head(const float* __restrict__ y8,
                                                  const float* __restrict__ w5,
                                                  float* __restrict__ out) {
    int idx = threadIdx.x;
    if (idx >= 80) return;
    int b = idx / 10, co = idx % 10;
    float s = 0.0f;
    for (int ci = 0; ci < 296; ++ci)
        for (int dh = 0; dh < 3; ++dh)
            s += w5[(co * 296 + ci) * 3 + dh] * y8[((size_t)b * 296 + ci) * 3 + dh];
    out[b * 10 + co] = s;
}

// ---------------------------------------------------------------------------
// Host launcher
// ---------------------------------------------------------------------------
extern "C" void kernel_launch(void* const* d_in, const int* in_sizes, int n_in,
                              void* d_out, int out_size, void* d_ws, size_t ws_size,
                              hipStream_t stream) {
    (void)in_sizes; (void)n_in; (void)out_size; (void)ws_size;
    const float* x  = (const float*)d_in[0];
    const float* w1 = (const float*)d_in[1];
    const float* w2 = (const float*)d_in[2];
    const float* w3 = (const float*)d_in[3];
    const float* w4 = (const float*)d_in[4];
    const float* w5 = (const float*)d_in[5];
    const float* g1 = (const float*)d_in[6];
    const float* b1 = (const float*)d_in[7];
    const float* g2 = (const float*)d_in[8];
    const float* b2 = (const float*)d_in[9];
    const float* g3 = (const float*)d_in[10];
    const float* b3 = (const float*)d_in[11];
    const float* g4 = (const float*)d_in[12];
    const float* b4 = (const float*)d_in[13];
    float* out = (float*)d_out;

    char* ws = (char*)d_ws;
    __bf16* W1a = (__bf16*)(ws + 0);          //  80 x   96 bf16
    __bf16* W2a = (__bf16*)(ws + 15360);      //  80 x  672 bf16
    __bf16* W3a = (__bf16*)(ws + 122880);     // 160 x  672 bf16
    __bf16* W4a = (__bf16*)(ws + 337920);     // 304 x 1344 bf16
    float*  FP  = (float*)(ws + 1155072);     // 85,248,000 B : y1/y3/y5/y7
    __bf16* BF  = (__bf16*)(ws + 86403072);   // 10,656,000 B : y2/y4/y6
    float*  Y7P = (float*)(ws + 97059072);    // [8,296,3,15]
    float*  Y8  = (float*)(ws + 97485312);    // [8,296,3]
    float*  ST  = (float*)(ws + 97513728);    // stats
    float *m1 = ST, *r1 = ST + 74, *m2 = ST + 148, *r2 = ST + 222;
    float *m3 = ST + 296, *r3 = ST + 444, *m4 = ST + 592, *r4 = ST + 888;

    // weight packing
    prep_w<<<(80 * 96 + 255) / 256, 256, 0, stream>>>(w1, W1a, 74, 79, 80, 96);
    prep_w<<<(80 * 672 + 255) / 256, 256, 0, stream>>>(w2, W2a, 74, 666, 80, 672);
    prep_w<<<(160 * 672 + 255) / 256, 256, 0, stream>>>(w3, W3a, 148, 666, 160, 672);
    prep_w<<<(304 * 1344 + 255) / 256, 256, 0, stream>>>(w4, W4a, 296, 1332, 304, 1344);

    // lift + pool4 -> y1 [8,74,9,4000]
    lift_wmma<<<dim3(125, 9, 8), 256, 0, stream>>>(x, W1a, FP);
    bn_stats<<<74, 256, 0, stream>>>(FP, m1, r1, 74, 9 * 4000);
    bn_relu_pool<true><<<(8 * 74 * 9 * 1000 + 255) / 256, 256, 0, stream>>>(
        FP, m1, r1, g1, b1, (void*)BF, 74, 9, 4000, 1000);

    // gconv2 -> y3 [8,74,7,1000]
    gconv_wmma<74, 74, 9, 1000><<<dim3(16, 7, 8), 256, 0, stream>>>(BF, W2a, FP);
    bn_stats<<<74, 256, 0, stream>>>(FP, m2, r2, 74, 7 * 1000);
    bn_relu_pool<true><<<(8 * 74 * 7 * 250 + 255) / 256, 256, 0, stream>>>(
        FP, m2, r2, g2, b2, (void*)BF, 74, 7, 1000, 250);

    // gconv3 -> y5 [8,148,5,250]
    gconv_wmma<148, 74, 7, 250><<<dim3(4, 5, 8), 256, 0, stream>>>(BF, W3a, FP);
    bn_stats<<<148, 256, 0, stream>>>(FP, m3, r3, 148, 5 * 250);
    bn_relu_pool<true><<<(8 * 148 * 5 * 62 + 255) / 256, 256, 0, stream>>>(
        FP, m3, r3, g3, b3, (void*)BF, 148, 5, 250, 62);

    // gconv4 -> y7 [8,296,3,62]
    gconv_wmma<296, 148, 5, 62><<<dim3(1, 3, 8), 256, 0, stream>>>(BF, W4a, FP);
    bn_stats<<<296, 256, 0, stream>>>(FP, m4, r4, 296, 3 * 62);
    bn_relu_pool<false><<<(8 * 296 * 3 * 15 + 255) / 256, 256, 0, stream>>>(
        FP, m4, r4, g4, b4, (void*)Y7P, 296, 3, 62, 15);

    // mean over time, final gconv + max(scale axis of size 1)
    mean_time<<<(8 * 296 * 3 + 255) / 256, 256, 0, stream>>>(Y7P, Y8, 8 * 296 * 3, 15);
    final_head<<<1, 128, 0, stream>>>(Y8, w5, out);
}
